// base_MAE_75548474737131
// MI455X (gfx1250) — compile-verified, matrix operand order checked by reading, not run on hardware
//
#include <hip/hip_runtime.h>
#include <math.h>

#define N_NODES 32768
#define N_EDGES 262144
#define C_CH    8
#define CE_CH   4
#define K_NB    8
#define H_DIM   32
#define INV_DIM 5
#define EPSV    1e-6f

#define WR_STRIDE 36   // padded row stride (halves) for Wr in LDS: 72B -> bank stride 18
#define R_STRIDE  68   // padded row stride (floats) for R slab: 272B, 16B-aligned, conflict-free

typedef __attribute__((ext_vector_type(16))) _Float16 v16h;
typedef __attribute__((ext_vector_type(8)))  float    v8f;

__device__ __forceinline__ float fast_rcp(float x) { return __builtin_amdgcn_rcpf(x); }
__device__ __forceinline__ float silu_f(float x) { return x * fast_rcp(1.0f + __expf(-x)); }
__device__ __forceinline__ float sigm_f(float x) { return fast_rcp(1.0f + __expf(-x)); }

// ---------------- init: f0 <- node0, f1/a0/a1 <- 0 ----------------
__global__ void init_kernel(const float* __restrict__ node0,
                            float* __restrict__ f0, float* __restrict__ f1,
                            float* __restrict__ a0, float* __restrict__ a1) {
  int i = blockIdx.x * blockDim.x + threadIdx.x;
  if (i < N_NODES * C_CH) { f0[i] = node0[i]; a0[i] = 0.0f; }
  if (i < N_NODES * C_CH * 3) { f1[i] = 0.0f; a1[i] = 0.0f; }
}

__global__ void zero_acc_kernel(float* __restrict__ a0, float* __restrict__ a1) {
  int i = blockIdx.x * blockDim.x + threadIdx.x;
  if (i < N_NODES * C_CH) a0[i] = 0.0f;
  if (i < N_NODES * C_CH * 3) a1[i] = 0.0f;
}

// ---------------- fused edge kernel (WMMA f16 for R = h @ Wr^T) ----------------
// Block = 256 threads = 8 waves; each wave owns 16 edges.
__global__ __launch_bounds__(256) void edge_kernel(
    const float* __restrict__ rel_pos, const float* __restrict__ edge0,
    const int* __restrict__ src, const int* __restrict__ dst,
    const float* __restrict__ f0, const float* __restrict__ f1,
    const float* __restrict__ W1, const float* __restrict__ b1,
    const float* __restrict__ Wr, const float* __restrict__ br,
    float* __restrict__ a0, float* __restrict__ a1)
{
  __shared__ _Float16 WrLds[320 * WR_STRIDE];       // ~22.5 KB, padded rows
  __shared__ float    Rbuf[8][16 * R_STRIDE];       // ~34.8 KB, padded rows

  __builtin_prefetch(Wr, 0, 1);                     // global_prefetch_b8
  for (int i = threadIdx.x; i < 320 * H_DIM; i += 256) {
    int row = i >> 5, col = i & 31;
    WrLds[row * WR_STRIDE + col] = (_Float16)Wr[i];
  }
  __syncthreads();

  const int lane = threadIdx.x & 31;
  const int wave = threadIdx.x >> 5;
  const int g    = lane >> 4;               // lane half-group (K-half selector)
  const int eloc = lane & 15;               // edge row within wave tile
  const int ncol = lane & 15;               // output column within 16-wide tile
  const int e    = blockIdx.x * 128 + wave * 16 + eloc;

  // --- edge invariants + unit direction ---
  float rx = rel_pos[e * 3 + 0], ry = rel_pos[e * 3 + 1], rz = rel_pos[e * 3 + 2];
  float r  = sqrtf(rx * rx + ry * ry + rz * rz);
  float rn = fast_rcp(r + EPSV);
  float ux = rx * rn, uy = ry * rn, uz = rz * rn;
  float invf[INV_DIM];
  invf[0] = r;
  invf[1] = edge0[e * CE_CH + 0];
  invf[2] = edge0[e * CE_CH + 1];
  invf[3] = edge0[e * CE_CH + 2];
  invf[4] = edge0[e * CE_CH + 3];

  // --- A fragment: h = silu(inv @ W1^T + b1), 16x32 f16 WMMA layout ---
  v16h afrag;
#pragma unroll
  for (int i = 0; i < 16; ++i) {
    int j = i >> 1, p = i & 1;
    int k = (j < 4 ? j * 2 + p : 16 + (j - 4) * 2 + p) + 8 * g;
    float acc = b1[k];
#pragma unroll
    for (int jj = 0; jj < INV_DIM; ++jj) acc += invf[jj] * W1[k * INV_DIM + jj];
    afrag[i] = (_Float16)silu_f(acc);
  }

  // --- gather source-node features for this lane's edge ---
  int sN = src[e];
  float f0s[8], f1s[8][3], dotv[8], crs[8][3];
#pragma unroll
  for (int c = 0; c < 8; ++c) {
    f0s[c] = f0[sN * 8 + c];
#pragma unroll
    for (int d = 0; d < 3; ++d) f1s[c][d] = f1[(sN * 8 + c) * 3 + d];
    dotv[c]   = f1s[c][0] * ux + f1s[c][1] * uy + f1s[c][2] * uz;
    crs[c][0] = uy * f1s[c][2] - uz * f1s[c][1];
    crs[c][1] = uz * f1s[c][0] - ux * f1s[c][2];
    crs[c][2] = ux * f1s[c][1] - uy * f1s[c][0];
  }

  float m0[4]    = {0.0f, 0.0f, 0.0f, 0.0f};
  float m1[4][3] = {};

  float* myR = Rbuf[wave];

  for (int p = 0; p < 5; ++p) {
    // -------- produce: 4 WMMA tiles = one TFN path (64 outputs x 16 edges) --------
#pragma unroll
    for (int t = 0; t < 4; ++t) {
      const int tile = p * 4 + t;
      v16h bfrag;
#pragma unroll
      for (int i = 0; i < 16; ++i) {
        int j = i >> 1, q = i & 1;
        int k = (j < 4 ? j * 2 + q : 16 + (j - 4) * 2 + q) + 8 * g;
        bfrag[i] = WrLds[(tile * 16 + ncol) * WR_STRIDE + k]; // B[k][col] = Wr[col][k]
      }
      v8f acc = {};
      acc = __builtin_amdgcn_wmma_f32_16x16x32_f16(false, afrag, false, bfrag,
                                                   (short)0, acc, false, false);
      float bias = br[tile * 16 + ncol];
#pragma unroll
      for (int v = 0; v < 8; ++v)                 // C layout: row M = v + 8*g
        myR[(v + 8 * g) * R_STRIDE + t * 16 + ncol] = acc[v] + bias;
    }
    __syncthreads();

    // -------- consume: lane handles edge eloc, output channels g*4 .. g*4+3 --------
#pragma unroll
    for (int o4 = 0; o4 < 4; ++o4) {
      const int o = g * 4 + o4;
      const float* Re = &myR[eloc * R_STRIDE + o * 8];
      if (p == 0) {
        float s = 0.0f;
#pragma unroll
        for (int c = 0; c < 8; ++c) s += Re[c] * f0s[c];
        m0[o4] += s;
      } else if (p == 1) {
        float s = 0.0f;
#pragma unroll
        for (int c = 0; c < 8; ++c) s += Re[c] * dotv[c];
        m0[o4] += s;
      } else if (p == 2) {
        float s = 0.0f;
#pragma unroll
        for (int c = 0; c < 8; ++c) s += Re[c] * f0s[c];
        m1[o4][0] += s * ux; m1[o4][1] += s * uy; m1[o4][2] += s * uz;
      } else if (p == 3) {
#pragma unroll
        for (int d = 0; d < 3; ++d) {
          float s = 0.0f;
#pragma unroll
          for (int c = 0; c < 8; ++c) s += Re[c] * f1s[c][d];
          m1[o4][d] += s;
        }
      } else {
#pragma unroll
        for (int d = 0; d < 3; ++d) {
          float s = 0.0f;
#pragma unroll
          for (int c = 0; c < 8; ++c) s += Re[c] * crs[c][d];
          m1[o4][d] += s;
        }
      }
    }
    __syncthreads();
  }

  // -------- scatter-add messages to destination node --------
  int dN = dst[e];
#pragma unroll
  for (int o4 = 0; o4 < 4; ++o4) {
    const int o = g * 4 + o4;
    atomicAdd(&a0[dN * 8 + o], m0[o4]);
#pragma unroll
    for (int d = 0; d < 3; ++d)
      atomicAdd(&a1[(dN * 8 + o) * 3 + d], m1[o4][d]);
  }
}

// ---------------- per-node update ----------------
__global__ void node_kernel(const float* __restrict__ self0, const float* __restrict__ gate_w,
                            const float* __restrict__ gate_b, const float* __restrict__ self1,
                            const float* __restrict__ a0, const float* __restrict__ a1,
                            float* __restrict__ f0, float* __restrict__ f1)
{
  int nd = blockIdx.x * blockDim.x + threadIdx.x;
  if (nd >= N_NODES) return;
  float f0l[8], nf0[8], gate[8], t1[8][3];
#pragma unroll
  for (int c = 0; c < 8; ++c) f0l[c] = f0[nd * 8 + c];
#pragma unroll
  for (int o = 0; o < 8; ++o) {
    float s = a0[nd * 8 + o];
#pragma unroll
    for (int c = 0; c < 8; ++c) s += self0[o * 8 + c] * f0l[c];
    nf0[o] = silu_f(s);
  }
#pragma unroll
  for (int o = 0; o < 8; ++o) {
    float s = gate_b[o];
#pragma unroll
    for (int c = 0; c < 8; ++c) s += gate_w[o * 8 + c] * nf0[c];
    gate[o] = sigm_f(s);
  }
#pragma unroll
  for (int c = 0; c < 8; ++c)
#pragma unroll
    for (int d = 0; d < 3; ++d)
      t1[c][d] = f1[(nd * 8 + c) * 3 + d] + a1[(nd * 8 + c) * 3 + d];
#pragma unroll
  for (int o = 0; o < 8; ++o) {
#pragma unroll
    for (int d = 0; d < 3; ++d) {
      float s = 0.0f;
#pragma unroll
      for (int c = 0; c < 8; ++c) s += self1[o * 8 + c] * t1[c][d];
      f1[(nd * 8 + o) * 3 + d] = s * gate[o];
    }
    f0[nd * 8 + o] = nf0[o];
  }
}

// ---------------- output projection: pos = Wout1 @ f1 (one thread per node) ----------------
__global__ void pos_kernel(const float* __restrict__ Wout1,
                           const float* __restrict__ f1, float* __restrict__ pos) {
  int nd = blockIdx.x * blockDim.x + threadIdx.x;
  if (nd >= N_NODES) return;
#pragma unroll
  for (int d = 0; d < 3; ++d) {
    float s = 0.0f;
#pragma unroll
    for (int c = 0; c < 8; ++c) s += Wout1[c] * f1[(nd * 8 + c) * 3 + d];
    pos[nd * 3 + d] = s;
  }
}

// ---------------- local geometry stats ----------------
__global__ void stats_kernel(const int* __restrict__ neighbors,
                             const float* __restrict__ nmask,
                             const float* __restrict__ pos,
                             float* __restrict__ out_len, float* __restrict__ out_ang) {
  int nd = blockIdx.x * blockDim.x + threadIdx.x;
  if (nd >= N_NODES) return;
  float px = pos[nd * 3 + 0], py = pos[nd * 3 + 1], pz = pos[nd * 3 + 2];
  float un[K_NB][3], mk[K_NB];
#pragma unroll
  for (int k = 0; k < K_NB; ++k) {
    int nb = neighbors[nd * K_NB + k];
    float vx = pos[nb * 3 + 0] - px;
    float vy = pos[nb * 3 + 1] - py;
    float vz = pos[nb * 3 + 2] - pz;
    float ln = sqrtf(vx * vx + vy * vy + vz * vz);
    mk[k] = nmask[nd * K_NB + k];
    out_len[nd * K_NB + k] = ln * mk[k];
    float s = fast_rcp(ln + EPSV);
    un[k][0] = vx * s; un[k][1] = vy * s; un[k][2] = vz * s;
  }
#pragma unroll
  for (int k = 0; k < K_NB; ++k)
#pragma unroll
    for (int l = 0; l < K_NB; ++l) {
      float cs = un[k][0] * un[l][0] + un[k][1] * un[l][1] + un[k][2] * un[l][2];
      cs = fminf(0.999f, fmaxf(-0.999f, cs));
      out_ang[(nd * K_NB + k) * K_NB + l] = acosf(cs) * mk[k] * mk[l];
    }
}

// ---------------- host launch ----------------
extern "C" void kernel_launch(void* const* d_in, const int* in_sizes, int n_in,
                              void* d_out, int out_size, void* d_ws, size_t ws_size,
                              hipStream_t stream)
{
  (void)in_sizes; (void)n_in; (void)out_size; (void)ws_size;
  const float* node0     = (const float*)d_in[0];
  const float* edge0     = (const float*)d_in[1];
  const float* rel_pos   = (const float*)d_in[2];
  const int*   src       = (const int*)d_in[3];
  const int*   dst       = (const int*)d_in[4];
  const int*   neighbors = (const int*)d_in[5];
  const float* nmask     = (const float*)d_in[6];
  const float* Wout1     = (const float*)d_in[7];
  // params pytree order per layer: W1, Wr, b1, br, gate_b, gate_w, self0, self1

  float* f0  = (float*)d_ws;
  float* f1  = f0 + N_NODES * C_CH;
  float* a0  = f1 + N_NODES * C_CH * 3;
  float* a1  = a0 + N_NODES * C_CH;
  float* pos = a1 + N_NODES * C_CH * 3;

  init_kernel<<<(N_NODES * 24 + 255) / 256, 256, 0, stream>>>(node0, f0, f1, a0, a1);

  for (int l = 0; l < 2; ++l) {
    const float* W1     = (const float*)d_in[8 + l * 8 + 0];
    const float* Wr     = (const float*)d_in[8 + l * 8 + 1];
    const float* b1     = (const float*)d_in[8 + l * 8 + 2];
    const float* br     = (const float*)d_in[8 + l * 8 + 3];
    const float* gate_b = (const float*)d_in[8 + l * 8 + 4];
    const float* gate_w = (const float*)d_in[8 + l * 8 + 5];
    const float* self0  = (const float*)d_in[8 + l * 8 + 6];
    const float* self1  = (const float*)d_in[8 + l * 8 + 7];
    if (l > 0)
      zero_acc_kernel<<<(N_NODES * 24 + 255) / 256, 256, 0, stream>>>(a0, a1);
    edge_kernel<<<N_EDGES / 128, 256, 0, stream>>>(rel_pos, edge0, src, dst, f0, f1,
                                                   W1, b1, Wr, br, a0, a1);
    node_kernel<<<(N_NODES + 255) / 256, 256, 0, stream>>>(self0, gate_w, gate_b, self1,
                                                           a0, a1, f0, f1);
  }

  pos_kernel<<<(N_NODES + 255) / 256, 256, 0, stream>>>(Wout1, f1, pos);

  float* out_len = (float*)d_out;
  float* out_ang = out_len + N_NODES * K_NB;
  stats_kernel<<<(N_NODES + 255) / 256, 256, 0, stream>>>(neighbors, nmask, pos,
                                                          out_len, out_ang);
}
